// memory_saver_56075093017369
// MI455X (gfx1250) — compile-verified, matrix operand order checked by reading, not run on hardware
//
#include <hip/hip_runtime.h>
#include <math.h>

// ---------------- problem constants ----------------
constexpr int Bb    = 2;
constexpr int Hh    = 32;
constexpr int BH    = 64;           // Bb*Hh
constexpr int S     = 8192;
constexpr int D     = 128;
constexpr int LOCAL = 64;
constexpr int KEEP  = 256;          // KEEP_HH
constexpr int KTOP  = KEEP - LOCAL; // 192
constexpr int MC    = 256;          // MEM_COMPRESS
constexpr int SP    = S - KEEP;     // 7936
constexpr int SLOC  = S - LOCAL;    // 8128
constexpr int TT    = 64;           // sp tile width
constexpr int NCH   = SP / TT;      // 124

// ---------------- workspace layout (bytes) ----------------
constexpr size_t OFF_HH   = 0;                       // BH*KEEP int      (64 KB)
constexpr size_t OFF_NON  = 65536;                   // BH*SP int        (~2 MB)
constexpr size_t OFF_RMAP = 2097152;                 // BH*MC int        (64 KB)
constexpr size_t OFF_W0H  = 2162688;                 // 3*512*256 f16    (768 KB)
constexpr size_t OFF_W1H  = 2949120;                 // 3*256*512 f16    (768 KB)
constexpr size_t OFF_CM   = 3735552;                 // BH*256*NCH f32   (~8 MB)
constexpr size_t OFF_RM   = 11862016;                // BH*256 f32
constexpr size_t OFF_LG   = 11927552;                // BH*256*SP f16    (~260 MB)

// ---------------- types ----------------
typedef _Float16 h16 __attribute__((ext_vector_type(16)));
typedef _Float16 h8  __attribute__((ext_vector_type(8)));
typedef float    f8  __attribute__((ext_vector_type(8)));
typedef unsigned int u32x4 __attribute__((ext_vector_type(4)));
typedef int i32x4 __attribute__((ext_vector_type(4)));
typedef int i32x8 __attribute__((ext_vector_type(8)));

#define HAVE_TDM __has_builtin(__builtin_amdgcn_tensor_load_to_lds)

// ---------------- helpers ----------------
__device__ __forceinline__ unsigned f2k(float f) {
  unsigned u = __float_as_uint(f);
  return (u & 0x80000000u) ? ~u : (u | 0x80000000u);
}
__device__ __forceinline__ float kdec(unsigned v) {
  return (v & 0x80000000u) ? __uint_as_float(v & 0x7FFFFFFFu) : __uint_as_float(~v);
}
__device__ __forceinline__ float siluf(float v) { return v / (1.0f + __expf(-v)); }

// Load one 16-half WMMA fragment per the CDNA5 16-bit A/B layout:
// lane half hl selects K in {kb + hl*8 .. +7} and {kb+16 + hl*8 .. +7}
__device__ __forceinline__ h16 frag(const _Float16* row, int kbase, int hl) {
  h8 lo = *(const h8*)(row + kbase + hl * 8);
  h8 hi = *(const h8*)(row + kbase + 16 + hl * 8);
  h16 r;
#pragma unroll
  for (int j = 0; j < 8; ++j) { r[j] = lo[j]; r[8 + j] = hi[j]; }
  return r;
}

__device__ __forceinline__ void block_scan256(int* sbuf, int tid) {
#pragma unroll
  for (int off = 1; off < 256; off <<= 1) {
    int v = (tid >= off) ? sbuf[tid - off] : 0;
    __syncthreads();
    sbuf[tid] += v;
    __syncthreads();
  }
}

// async copy 16B global -> LDS (ASYNCcnt path, gfx1250)
__device__ __forceinline__ void async_g2l_b128(void* lds_dst, const void* gbase, unsigned byte_off) {
  unsigned ldsa = (unsigned)(size_t)lds_dst;
  asm volatile("global_load_async_to_lds_b128 %0, %1, %2"
               :: "v"(ldsa), "v"(byte_off), "s"(gbase)
               : "memory");
}
__device__ __forceinline__ void wait_async0() {
  asm volatile("s_wait_asynccnt 0x0" ::: "memory");
}

// TDM: DMA a 2D f16 tile (tile_d0 x tile_d1, row stride stride0, dims in elements)
// from global to LDS (tile stored contiguously). Issue from ONE wave per workgroup.
__device__ __forceinline__ void tdm_load_2d_f16(unsigned lds_addr, const void* gtile,
                                                unsigned tensor_d0, unsigned tensor_d1,
                                                unsigned tile_d0, unsigned tile_d1,
                                                unsigned stride0) {
#if HAVE_TDM
  unsigned long long ga = (unsigned long long)(size_t)gtile;
  // D# group0: count=1, lds_addr, global_addr[56:0], type=2 ("image")
  u32x4 g0 = { 1u,
               lds_addr,
               (unsigned)(ga & 0xFFFFFFFFu),
               ((unsigned)((ga >> 32) & 0x01FFFFFFu)) | (2u << 30) };
  // D# group1: data_size=1 (2B); tensor_dim0/1, tile_dim0/1, tensor_dim0_stride
  i32x8 g1 = { (int)(1u << 16),
               (int)(tensor_d0 << 16),
               (int)((tensor_d0 >> 16) | (tensor_d1 << 16)),
               (int)((tensor_d1 >> 16) | (tile_d0 << 16)),
               (int)(tile_d1 & 0xFFFFu),
               (int)stride0,
               0, 0 };
  i32x4 z4 = { 0, 0, 0, 0 };
#if __clang_major__ >= 23
  i32x8 z8 = { 0, 0, 0, 0, 0, 0, 0, 0 };
  __builtin_amdgcn_tensor_load_to_lds(g0, g1, z4, z4, z8, 0);
#else
  __builtin_amdgcn_tensor_load_to_lds(g0, g1, z4, z4, 0);
#endif
#else
  (void)lds_addr; (void)gtile; (void)tensor_d0; (void)tensor_d1;
  (void)tile_d0; (void)tile_d1; (void)stride0;
#endif
}
__device__ __forceinline__ void wait_tensor0() {
#if HAVE_TDM
  __builtin_amdgcn_s_wait_tensorcnt(0);
#endif
}

// ================= kernel 1: selection (radix top-k, stable partition) =================
__global__ __launch_bounds__(256) void k_select(const float* __restrict__ sc_all,
                                                int* __restrict__ hh_pos,
                                                int* __restrict__ non_pos,
                                                int* __restrict__ rmap) {
  __shared__ unsigned hist[256];
  __shared__ int sbuf[256];
  __shared__ unsigned sh_dig;
  __shared__ int sh_c;
  __shared__ unsigned char m8[S];

  const int bh = blockIdx.x, tid = threadIdx.x;
  const float* sc = sc_all + (size_t)bh * S;
  int* hhp = hh_pos + bh * KEEP;
  int* nnp = non_pos + bh * SP;
  int* rmp = rmap + bh * MC;

  // ---- pass A: kth-largest (k=KTOP) over sc[0:SLOC) ----
  unsigned prefix = 0; int kk = KTOP;
  for (int pass = 0; pass < 4; ++pass) {
    const int shift = 24 - pass * 8;
    const unsigned maskHi = pass ? (0xFFFFFFFFu << (32 - 8 * pass)) : 0u;
    hist[tid] = 0u;
    __syncthreads();
    for (int i = tid; i < SLOC; i += 256) {
      unsigned key = f2k(sc[i]);
      if ((key & maskHi) == prefix) atomicAdd(&hist[(key >> shift) & 255], 1u);
    }
    __syncthreads();
    if (tid == 0) {
      unsigned c = 0; int d = 255;
      for (; d > 0; --d) { unsigned h = hist[d]; if (c + h >= (unsigned)kk) break; c += h; }
      sh_dig = (unsigned)d; sh_c = (int)c;
    }
    __syncthreads();
    prefix |= (sh_dig << shift);
    kk -= sh_c;
    __syncthreads();
  }
  const unsigned thr = prefix;
  const int ties = kk;

  // ---- keep-mask with stable (lowest-index-first) tie selection ----
  int tiec = 0;
  for (int base = 0; base < S; base += 256) {
    int i = base + tid;
    int gt = 0, eq = 0;
    if (i < SLOC) { unsigned key = f2k(sc[i]); gt = key > thr; eq = key == thr; }
    sbuf[tid] = eq; __syncthreads();
    block_scan256(sbuf, tid);
    int excl = sbuf[tid] - eq;
    int take = eq && (tiec + excl < ties);
    m8[i] = (unsigned char)((i >= SLOC) ? 1 : (gt || take));
    tiec += sbuf[255];
    __syncthreads();
  }

  // ---- stable partition ----
  int ck = 0, cn = 0;
  for (int base = 0; base < S; base += 256) {
    int i = base + tid;
    int kept = m8[i];
    sbuf[tid] = kept; __syncthreads();
    block_scan256(sbuf, tid);
    int incl = sbuf[tid];
    if (kept) hhp[ck + incl - 1] = i;
    else      nnp[cn + (tid - incl)] = i;
    ck += sbuf[255];
    cn += 256 - sbuf[255];
    __syncthreads();
  }

  // ---- pass B: top-MC over non_scores ----
  prefix = 0; kk = MC;
  for (int pass = 0; pass < 4; ++pass) {
    const int shift = 24 - pass * 8;
    const unsigned maskHi = pass ? (0xFFFFFFFFu << (32 - 8 * pass)) : 0u;
    hist[tid] = 0u;
    __syncthreads();
    for (int j = tid; j < SP; j += 256) {
      unsigned key = f2k(sc[nnp[j]]);
      if ((key & maskHi) == prefix) atomicAdd(&hist[(key >> shift) & 255], 1u);
    }
    __syncthreads();
    if (tid == 0) {
      unsigned c = 0; int d = 255;
      for (; d > 0; --d) { unsigned h = hist[d]; if (c + h >= (unsigned)kk) break; c += h; }
      sh_dig = (unsigned)d; sh_c = (int)c;
    }
    __syncthreads();
    prefix |= (sh_dig << shift);
    kk -= sh_c;
    __syncthreads();
  }
  const unsigned thr2 = prefix;
  const int ties2 = kk;

  int cnt = 0; tiec = 0;
  for (int base = 0; base < SP; base += 256) {
    int j = base + tid;
    unsigned key = f2k(sc[nnp[j]]);
    int gt = key > thr2, eq = key == thr2;
    sbuf[tid] = eq; __syncthreads();
    block_scan256(sbuf, tid);
    int exclEq = sbuf[tid] - eq;
    int totEq = sbuf[255];
    int sel = gt || (eq && (tiec + exclEq < ties2));
    __syncthreads();
    sbuf[tid] = sel; __syncthreads();
    block_scan256(sbuf, tid);
    int exclS = sbuf[tid] - sel;
    if (sel) rmp[cnt + exclS] = j;
    cnt += sbuf[255];
    tiec += totEq;
    __syncthreads();
  }
}

// ================= kernel 2: gather heavy-hitter rows straight to output =================
__global__ __launch_bounds__(128) void k_gather_hh(const float* __restrict__ pk,
                                                   const float* __restrict__ pv,
                                                   const int* __restrict__ hh_pos,
                                                   float* __restrict__ out) {
  const int bh = blockIdx.y, i = blockIdx.x, t = threadIdx.x;
  const int s = hh_pos[bh * KEEP + i];
  const size_t OUTK = (size_t)BH * 512 * D;
  const size_t src = ((size_t)bh * S + s) * D + t;
  const size_t dst = ((size_t)bh * 512 + i) * D + t;
  out[dst] = pk[src];
  out[OUTK + dst] = pv[src];
}

// ================= kernel 3: weights -> f16, tap-major =================
__global__ __launch_bounds__(256) void k_wcvt(const float* __restrict__ w0,
                                              const float* __restrict__ w1,
                                              _Float16* __restrict__ W0h,
                                              _Float16* __restrict__ W1h) {
  const int n0 = 512 * 256 * 3;
  const int n1 = 256 * 512 * 3;
  int idx = blockIdx.x * 256 + threadIdx.x;
  if (idx < n0) {
    int t = idx % 3, c = (idx / 3) % 256, m = idx / (3 * 256);
    W0h[((size_t)t * 512 + m) * 256 + c] = (_Float16)w0[idx];
  } else if (idx < n0 + n1) {
    int j = idx - n0;
    int t = j % 3, c = (j / 3) % 512, m = j / (3 * 512);
    W1h[((size_t)t * 256 + m) * 512 + c] = (_Float16)w1[j];
  }
}

// ================= kernel 4: fused conv0+silu+conv1+silu (TDM/async staged slabs) =========
__global__ __launch_bounds__(256) void k_conv(const float* __restrict__ pk,
                                              const float* __restrict__ pv,
                                              const int* __restrict__ non_pos,
                                              const _Float16* __restrict__ W0h,
                                              const _Float16* __restrict__ W1h,
                                              const float* __restrict__ b0,
                                              const float* __restrict__ b1,
                                              _Float16* __restrict__ logits,
                                              float* __restrict__ chunkmax) {
  __shared__ _Float16 xin[84 * 256];    // [j][c]   s = ch0-2+j
  __shared__ _Float16 y0[68 * 512];     // [jy][m]  s = ch0-1+jy (jy<66 meaningful)
  __shared__ _Float16 wslab[512 * 32];  // weight k-slab (32 KB), reused by both GEMMs
  __shared__ unsigned cmax[256];

  const int ch = blockIdx.x, bh = blockIdx.y, tid = threadIdx.x;
  const int ch0 = ch * TT;
  const int* nnp = non_pos + bh * SP;

  // ---- gather input tile (coalesced over channels) ----
  for (int j = 0; j < 84; ++j) {
    int s = ch0 - 2 + j;
    float v = 0.0f;
    if (s >= 0 && s < SP) {
      int row = nnp[s];
      v = (tid < 128) ? pk[((size_t)bh * S + row) * D + tid]
                      : pv[((size_t)bh * S + row) * D + (tid - 128)];
    }
    xin[j * 256 + tid] = (_Float16)v;
  }
  cmax[tid] = 0u;
  __syncthreads();

  const int wave = tid >> 5, lane = tid & 31, l16 = lane & 15, hl = lane >> 4;

  // =========== GEMM0: y0 = silu(W0 x xin + b0), 3 taps, K=256 ===========
  // wave owns Mtiles {4w..4w+3} x 5 Ntiles, accumulators live in registers
  {
    f8 acc0[20];
#pragma unroll
    for (int i = 0; i < 20; ++i) acc0[i] = (f8){};

    for (int t = 0; t < 3; ++t) {
      for (int k2 = 0; k2 < 8; ++k2) {
        __syncthreads();  // previous slab consumers done
        const _Float16* wsrc = W0h + (size_t)t * 512 * 256 + k2 * 32;
#if HAVE_TDM
        // Tensor Data Mover: 512x32-half tile from a 512x256 tensor (stride 256)
        if (wave == 0) {
          tdm_load_2d_f16((unsigned)(size_t)&wslab[0], wsrc,
                          /*tensor_d0=*/256, /*tensor_d1=*/512,
                          /*tile_d0=*/32, /*tile_d1=*/512, /*stride0=*/256);
          wait_tensor0();
        }
#else
        for (int e = tid; e < 1024; e += 256) {
          int r = e >> 1, part = e & 1;
          *(uint4*)&wslab[r * 32 + part * 16] = *(const uint4*)(wsrc + r * 256 + part * 16);
        }
#endif
        if (!(t == 2 && k2 == 7)) {
          int nk = (k2 + 1) & 7, ntap = t + (k2 == 7);
          __builtin_prefetch(W0h + (size_t)ntap * 512 * 256 + nk * 32 + (tid & 127) * 4 * 256, 0, 0);
        }
        __syncthreads();

        h16 af[4];
#pragma unroll
        for (int i = 0; i < 4; ++i)
          af[i] = frag(wslab + ((wave * 4 + i) * 16 + l16) * 32, 0, hl);
#pragma unroll
        for (int nt = 0; nt < 5; ++nt) {
          h16 bf = frag(xin + (nt * 16 + l16 + t) * 256, k2 * 32, hl);
#pragma unroll
          for (int i = 0; i < 4; ++i)
            acc0[i * 5 + nt] = __builtin_amdgcn_wmma_f32_16x16x32_f16(
                false, af[i], false, bf, (short)0, acc0[i * 5 + nt], false, false);
        }
      }
    }
    // epilogue: bias + silu -> y0 (f16)
#pragma unroll
    for (int i = 0; i < 4; ++i) {
      int mt = wave * 4 + i;
#pragma unroll
      for (int nt = 0; nt < 5; ++nt) {
        int n = nt * 16 + l16;
#pragma unroll
        for (int r = 0; r < 8; ++r) {
          int M = mt * 16 + r + hl * 8;
          float v = siluf(acc0[i * 5 + nt][r] + b0[M]);
          if (n < 68) y0[n * 512 + M] = (_Float16)v;
        }
      }
    }
  }

  // =========== GEMM1: logits = silu(W1 x y0 + b1), 3 taps, K=512 ===========
  // wave owns Mtiles {w, w+8} x 4 Ntiles; slab = 256 rows x 64 halves (two k-steps)
  {
    f8 acc1[8];
#pragma unroll
    for (int i = 0; i < 8; ++i) acc1[i] = (f8){};

    for (int t = 0; t < 3; ++t) {
      for (int k4 = 0; k4 < 8; ++k4) {
        __syncthreads();  // y0 writes (first iter) / previous slab consumers done
        // async-stage (ASYNCcnt path): wslab[r][0:64] = W1h[t][r][k4*64 + 0:64]
        const unsigned gslab = (unsigned)(((size_t)t * 256 * 512 + k4 * 64) * sizeof(_Float16));
        for (int e = tid; e < 1024; e += 256) {
          int r = e >> 2, part = e & 3;
          async_g2l_b128(&wslab[r * 64 + part * 16], W1h,
                         gslab + (unsigned)((r * 512 + part * 16) * sizeof(_Float16)));
        }
        if (!(t == 2 && k4 == 7)) {
          int nk = (k4 + 1) & 7, ntap = t + (k4 == 7);
          __builtin_prefetch(W1h + (size_t)ntap * 256 * 512 + nk * 64 + (tid & 63) * 4 * 512, 0, 0);
        }
        wait_async0();
        __syncthreads();

        h16 a0a = frag(wslab + (wave * 16 + l16) * 64, 0, hl);
        h16 a0b = frag(wslab + (wave * 16 + l16) * 64, 32, hl);
        h16 a1a = frag(wslab + ((wave + 8) * 16 + l16) * 64, 0, hl);
        h16 a1b = frag(wslab + ((wave + 8) * 16 + l16) * 64, 32, hl);
#pragma unroll
        for (int nt = 0; nt < 4; ++nt) {
          const _Float16* brow = y0 + (nt * 16 + l16 + t) * 512;
          h16 bb0 = frag(brow, k4 * 64, hl);
          h16 bb1 = frag(brow, k4 * 64 + 32, hl);
          acc1[nt]     = __builtin_amdgcn_wmma_f32_16x16x32_f16(false, a0a, false, bb0, (short)0, acc1[nt],     false, false);
          acc1[nt]     = __builtin_amdgcn_wmma_f32_16x16x32_f16(false, a0b, false, bb1, (short)0, acc1[nt],     false, false);
          acc1[4 + nt] = __builtin_amdgcn_wmma_f32_16x16x32_f16(false, a1a, false, bb0, (short)0, acc1[4 + nt], false, false);
          acc1[4 + nt] = __builtin_amdgcn_wmma_f32_16x16x32_f16(false, a1b, false, bb1, (short)0, acc1[4 + nt], false, false);
        }
      }
    }
    // epilogue: bias + silu -> f16 logits + chunk max
#pragma unroll
    for (int i = 0; i < 2; ++i) {
      int mt = wave + i * 8;
#pragma unroll
      for (int nt = 0; nt < 4; ++nt) {
        int n = nt * 16 + l16;
        int scol = ch0 + n;
#pragma unroll
        for (int r = 0; r < 8; ++r) {
          int M = mt * 16 + r + hl * 8;
          float v = siluf(acc1[i * 4 + nt][r] + b1[M]);
          logits[((size_t)bh * 256 + M) * SP + scol] = (_Float16)v;
          atomicMax(&cmax[M], f2k(v));
        }
      }
    }
  }
  __syncthreads();
  chunkmax[((size_t)bh * 256 + tid) * NCH + ch] = kdec(cmax[tid]);
}

// ================= kernel 5: row max over chunks =================
__global__ __launch_bounds__(256) void k_rowmax(const float* __restrict__ chunkmax,
                                                float* __restrict__ rowmax) {
  const int bh = blockIdx.x, m = threadIdx.x;
  const float* p = chunkmax + ((size_t)bh * 256 + m) * NCH;
  float mx = -INFINITY;
  for (int c = 0; c < NCH; ++c) mx = fmaxf(mx, p[c]);
  rowmax[bh * 256 + m] = mx;
}

// ================= kernel 6: softmax-weighted merge (TDM logits, exp -> WMMA) =============
__global__ __launch_bounds__(512) void k_merge(const float* __restrict__ pk,
                                               const float* __restrict__ pv,
                                               const int* __restrict__ non_pos,
                                               const int* __restrict__ rmap,
                                               const _Float16* __restrict__ logits,
                                               const float* __restrict__ rowmax,
                                               const float* __restrict__ nrm_p,
                                               float* __restrict__ out) {
  __shared__ _Float16 lraw[256 * 64];  // raw logits tile (TDM destination)
  __shared__ _Float16 pbuf[256 * 64];  // A: [m][k]      exp(l - max)
  __shared__ _Float16 kvT[256 * 64];   // B^T: [n][k]    n: 0..127 K, 128..255 V
  __shared__ float smax[256];
  __shared__ float ssum[256];

  const int bh = blockIdx.x, tid = threadIdx.x;
  const int* nnp = non_pos + bh * SP;
  if (tid < 256) { smax[tid] = rowmax[bh * 256 + tid]; ssum[tid] = 0.0f; }
  __syncthreads();

  const int wave = tid >> 5, lane = tid & 31, l16 = lane & 15, hl = lane >> 4;
  f8 acc[16];
#pragma unroll
  for (int i = 0; i < 16; ++i) acc[i] = (f8){};

  for (int chn = 0; chn < NCH; ++chn) {
    // phase 1: issue TDM for the 256x64 logits tile, overlap with K/V gather
#if HAVE_TDM
    if (wave == 0) {
      const _Float16* gtile = logits + (size_t)bh * 256 * SP + chn * TT;
      tdm_load_2d_f16((unsigned)(size_t)&lraw[0], gtile,
                      /*tensor_d0=*/SP, /*tensor_d1=*/256,
                      /*tile_d0=*/TT, /*tile_d1=*/256, /*stride0=*/SP);
    }
#endif
    // gather K/V tile transposed -> kvT (overlaps the tensor DMA)
    {
      int k = tid & 63, cg = (tid >> 6) * 32;
      int row = nnp[chn * TT + k];
      const float* srcp = (cg < 128) ? pk + ((size_t)bh * S + row) * D + cg
                                     : pv + ((size_t)bh * S + row) * D + (cg - 128);
#pragma unroll
      for (int j = 0; j < 32; ++j) kvT[(cg + j) * 64 + k] = (_Float16)srcp[j];
    }
#if HAVE_TDM
    if (wave == 0) wait_tensor0();
    __syncthreads();
#endif
    // phase 2: exp tile -> pbuf, accumulate softmax denominator
    {
      int m = tid >> 1, cg = (tid & 1) * 32;
#if HAVE_TDM
      const _Float16* src = lraw + m * 64 + cg;
#else
      const _Float16* src = logits + ((size_t)bh * 256 + m) * SP + chn * TT + cg;
      if (chn + 1 < NCH) __builtin_prefetch(src + TT, 0, 0);
#endif
      float mx = smax[m], ls = 0.0f;
#pragma unroll
      for (int j = 0; j < 32; ++j) {
        float e = __expf((float)src[j] - mx);
        ls += e;
        pbuf[m * 64 + cg + j] = (_Float16)e;
      }
      atomicAdd(&ssum[m], ls);
    }
    __syncthreads();

    // phase 3: wave = Mtile; 16 Ntiles; K = 64 (2 k-steps)
    const _Float16* arow = pbuf + (wave * 16 + l16) * 64;
    h16 a0 = frag(arow, 0, hl);
    h16 a1 = frag(arow, 32, hl);
#pragma unroll
    for (int nt = 0; nt < 16; ++nt) {
      const _Float16* brow = kvT + (nt * 16 + l16) * 64;
      h16 bb0 = frag(brow, 0, hl);
      h16 bb1 = frag(brow, 32, hl);
      acc[nt] = __builtin_amdgcn_wmma_f32_16x16x32_f16(false, a0, false, bb0, (short)0, acc[nt], false, false);
      acc[nt] = __builtin_amdgcn_wmma_f32_16x16x32_f16(false, a1, false, bb1, (short)0, acc[nt], false, false);
    }
    __syncthreads();
  }

  // finalize: softmax*normalizer + (1-normalizer)*one_hot residual, write merged K/V
  const float nrm = nrm_p[0];
  const size_t OUTK = (size_t)BH * 512 * D;
#pragma unroll
  for (int nt = 0; nt < 16; ++nt) {
    int n = nt * 16 + l16;
#pragma unroll
    for (int r = 0; r < 8; ++r) {
      int M = wave * 16 + r + hl * 8;
      float v = acc[nt][r] * nrm / ssum[M];
      int rj = rmap[bh * MC + M];
      int rrow = nnp[rj];
      float resv = (n < 128) ? pk[((size_t)bh * S + rrow) * D + n]
                             : pv[((size_t)bh * S + rrow) * D + (n - 128)];
      v += (1.0f - nrm) * resv;
      size_t dst = ((size_t)bh * 512 + KEEP + M) * D + (n & 127);
      if (n < 128) out[dst] = v;
      else         out[OUTK + dst] = v;
    }
  }
}

// ================= host launcher =================
extern "C" void kernel_launch(void* const* d_in, const int* in_sizes, int n_in,
                              void* d_out, int out_size, void* d_ws, size_t ws_size,
                              hipStream_t stream) {
  (void)in_sizes; (void)n_in; (void)out_size; (void)ws_size;
  const float* pk  = (const float*)d_in[0];
  const float* pv  = (const float*)d_in[1];
  const float* hh  = (const float*)d_in[2];
  const float* w0  = (const float*)d_in[3];
  const float* b0  = (const float*)d_in[4];
  const float* w1  = (const float*)d_in[5];
  const float* b1  = (const float*)d_in[6];
  const float* nrm = (const float*)d_in[7];
  float* out = (float*)d_out;

  char* ws = (char*)d_ws;
  int*       hhp = (int*)(ws + OFF_HH);
  int*       nnp = (int*)(ws + OFF_NON);
  int*       rmp = (int*)(ws + OFF_RMAP);
  _Float16*  W0h = (_Float16*)(ws + OFF_W0H);
  _Float16*  W1h = (_Float16*)(ws + OFF_W1H);
  float*     cm  = (float*)(ws + OFF_CM);
  float*     rm  = (float*)(ws + OFF_RM);
  _Float16*  lg  = (_Float16*)(ws + OFF_LG);

  k_select<<<BH, 256, 0, stream>>>(hh, hhp, nnp, rmp);
  k_wcvt<<<(2 * 3 * 512 * 256 + 255) / 256, 256, 0, stream>>>(w0, w1, W0h, W1h);
  k_gather_hh<<<dim3(KEEP, BH), 128, 0, stream>>>(pk, pv, hhp, out);
  k_conv<<<dim3(NCH, BH), 256, 0, stream>>>(pk, pv, nnp, W0h, W1h, b0, b1, lg, cm);
  k_rowmax<<<BH, 256, 0, stream>>>(cm, rm);
  k_merge<<<BH, 512, 0, stream>>>(pk, pv, nnp, rmp, lg, rm, nrm, out);
}
